// LSTM_59081570123993
// MI455X (gfx1250) — matrix operand order repaired and checked by reading, not compiled, not run on hardware
//
#include <hip/hip_runtime.h>
#include <hip/hip_bf16.h>

// ---------------------------------------------------------------------------
// LSTM (T=256, B=64, D=HID=1024, L=2) for gfx1250 (MI455X).
//  * Wx/Wh/X converted to bf16 once per call (weights stay resident in the
//    192MB L2 across the 512 sequential layer-steps: 33.6MB bf16).
//  * Per (t,layer): fused gate GEMM [64 x 4096] = [64 x 2048]*[2048 x 4096]
//    via v_wmma_f32_16x16x32_bf16, K = concat(x_t, h_{t-1}).
//  * Grid = 64 blocks (one per 16-wide hidden slice), block = 256 thr
//    (8 waves). Wave w: m_tile = w>>1, gate-pair = w&1 -> 2 accumulators
//    sharing one A fragment; K staged in 64-wide chunks -> 4 WMMA per wave
//    per barrier.
//  * All 4 gates of a hidden slice land in one block, so the LSTM cell
//    update (sigmoid/tanh via v_exp_f32) fuses in-kernel through LDS.
//  * h state double-buffered in bf16; layer0's h buffer is layer1's x input.
// ---------------------------------------------------------------------------

#define T_STEPS 256
#define BATCH   64
#define DIM     1024
#define HID     1024
#define NLAYERS 2

typedef __attribute__((ext_vector_type(16))) __bf16         v16bf;
typedef __attribute__((ext_vector_type(8)))  float          v8f;
typedef __attribute__((ext_vector_type(8)))  unsigned short u16x8;

__device__ __forceinline__ unsigned short f32_to_bf16_rne(float f) {
    unsigned int u = __float_as_uint(f);
    unsigned int lsb = (u >> 16) & 1u;
    u += 0x7fffu + lsb;
    return (unsigned short)(u >> 16);
}

__device__ __forceinline__ float sigmoid_f(float x) {
    return 1.0f / (1.0f + __expf(-x));
}
__device__ __forceinline__ float tanh_f(float x) {
    float e = __expf(2.0f * x);
    return (e - 1.0f) / (e + 1.0f);
}

// ---------------------------------------------------------------------------
// fp32 -> bf16 conversion (grid-stride)
// ---------------------------------------------------------------------------
__global__ void cvt_f32_bf16(const float* __restrict__ s,
                             unsigned short* __restrict__ d, long n) {
    long i = (long)blockIdx.x * blockDim.x + threadIdx.x;
    long stride = (long)gridDim.x * blockDim.x;
    for (; i < n; i += stride) d[i] = f32_to_bf16_rne(s[i]);
}

// ---------------------------------------------------------------------------
// One LSTM layer-step. grid = 64, block = 256 (8 waves).
// Wave w: m_tile = w>>1 (0..3), gate-pair gp = w&1 (gates 2gp, 2gp+1).
// K-loop: 32 chunks of 64 (2 sub-steps of 32 per chunk, 2 gates each
// => 4 v_wmma per wave per __syncthreads()).
// ---------------------------------------------------------------------------
__global__ __launch_bounds__(256, 1)
void lstm_step(const unsigned short* __restrict__ xsrc,   // [B][1024] bf16
               const unsigned short* __restrict__ hprev,  // [B][1024] bf16
               unsigned short*       __restrict__ hnext,  // [B][1024] bf16
               const unsigned short* __restrict__ Wxl,    // [4H][D]  bf16
               const unsigned short* __restrict__ Whl,    // [4H][H]  bf16
               const float*          __restrict__ biasl,  // [4H]
               float*                __restrict__ Cst,    // [B][H] fp32 in/out
               float*                __restrict__ outp,   // [B][H] or null
               float*                __restrict__ Hf,     // [B][H] or null
               float*                __restrict__ Cf) {   // [B][H] or null
    // A chunk: 64 rows x 64 k, row stride padded 64 -> 72 (144B, 16B-aligned)
    __shared__ __align__(16) unsigned short Abuf[2][64 * 72];
    __shared__ float gbuf[4][64][16];   // [gate][batch][j]

    const int tid  = threadIdx.x;
    const int lane = tid & 31;
    const int wave = tid >> 5;          // 0..7
    const int gp   = wave & 1;          // gate pair: gates 2gp, 2gp+1
    const int mt   = wave >> 1;         // 0..3 batch tile
    const int half = lane >> 4;         // 0/1
    const int nl   = lane & 15;
    const int jg   = blockIdx.x * 16 + nl;     // hidden unit (column)

    // accumulators init with per-gate bias (same column across tile rows)
    v8f acc[2];
#pragma unroll
    for (int g = 0; g < 2; ++g) {
        const float bv = biasl[(gp * 2 + g) * 1024 + jg];
#pragma unroll
        for (int r = 0; r < 8; ++r) acc[g][r] = bv;
    }

    // cooperative stage of a 64x64 bf16 A-chunk (256 thr x 16 elem = 32B ea)
    auto stage = [&](int buf, int kbase) {
        const unsigned short* src;
        int kloc;
        if (kbase < 1024) { src = xsrc;  kloc = kbase; }
        else              { src = hprev; kloc = kbase - 1024; }
        const int e   = tid * 16;       // 256*16 = 4096 elements
        const int row = e >> 6;
        const int col = e & 63;
        const unsigned short* sp = &src[row * 1024 + kloc + col];
        u16x8 a0 = *(const u16x8*)sp;
        u16x8 a1 = *(const u16x8*)(sp + 8);
        *(u16x8*)&Abuf[buf][row * 72 + col]     = a0;
        *(u16x8*)&Abuf[buf][row * 72 + col + 8] = a1;
    };

    stage(0, 0);
    __syncthreads();

    for (int kk = 0; kk < 32; ++kk) {
        const int cur = kk & 1;
        if (kk + 1 < 32) stage(cur ^ 1, (kk + 1) * 64);

        // prefetch next chunk of this wave's weight rows into GL2
        if (kk + 1 < 32) {
            const int kb2 = (kk + 1) * 64;
            const int nrow0 = (gp * 2) * 1024 + jg;
            const unsigned short* wr2 =
                (kb2 < 1024) ? (Wxl + (size_t)nrow0 * 1024 + kb2)
                             : (Whl + (size_t)nrow0 * 1024 + (kb2 - 1024));
            __builtin_prefetch((const void*)wr2, 0, 1);
        }

#pragma unroll
        for (int ks = 0; ks < 2; ++ks) {
            // A fragment (16x32 bf16): lane = M row, two 8-elt K chunks
            union { v16bf v; u16x8 u[2]; } af;
            const unsigned short* ap =
                &Abuf[cur][(mt * 16 + nl) * 72 + ks * 32];
            af.u[0] = *(const u16x8*)(ap + half * 8);
            af.u[1] = *(const u16x8*)(ap + 16 + half * 8);

            const int kbase = kk * 64 + ks * 32;
#pragma unroll
            for (int g = 0; g < 2; ++g) {
                // B fragment (32x16): column = weight row, two 8-elt K chunks
                const int nrow = (gp * 2 + g) * 1024 + jg;
                const unsigned short* wr =
                    (kbase < 1024) ? (Wxl + (size_t)nrow * 1024 + kbase)
                                   : (Whl + (size_t)nrow * 1024 + (kbase - 1024));
                union { v16bf v; u16x8 u[2]; } bf;
                bf.u[0] = *(const u16x8*)(wr + half * 8);
                bf.u[1] = *(const u16x8*)(wr + 16 + half * 8);

                acc[g] = __builtin_amdgcn_wmma_f32_16x16x32_bf16(
                    false, af.v, false, bf.v, (short)0, acc[g], false, false);
            }
        }
        __syncthreads();
    }

    // stash gate pre-activations: VGPR r -> batch row (mt*16 + half*8 + r)
#pragma unroll
    for (int g = 0; g < 2; ++g)
#pragma unroll
        for (int r = 0; r < 8; ++r)
            gbuf[gp * 2 + g][mt * 16 + half * 8 + r][nl] = acc[g][r];
    __syncthreads();

    // fused LSTM cell update: 256 threads x 4 elements cover 64x16
#pragma unroll
    for (int q = 0; q < 4; ++q) {
        const int e    = tid * 4 + q;
        const int brow = e >> 4;
        const int j    = e & 15;
        const int jcol = blockIdx.x * 16 + j;

        const float ig = gbuf[0][brow][j];
        const float fg = gbuf[1][brow][j];
        const float og = gbuf[2][brow][j];
        const float cg = gbuf[3][brow][j];

        const size_t sidx = (size_t)brow * 1024 + jcol;
        const float cold = Cst[sidx];
        const float cn = sigmoid_f(fg) * cold + sigmoid_f(ig) * tanh_f(cg);
        const float h  = sigmoid_f(og) * tanh_f(cn);

        Cst[sidx]   = cn;
        hnext[sidx] = f32_to_bf16_rne(h);
        if (outp) outp[sidx] = h;
        if (Hf)   Hf[sidx]   = h;
        if (Cf)   Cf[sidx]   = cn;
    }
}

// ---------------------------------------------------------------------------
// Host launcher.
// d_in: X [T,B,D] f32, Wx [L,4H,D] f32, Wh [L,4H,H] f32, b [L,4H] f32
// d_out: outputs [T,B,H] ++ Hf [L,B,H] ++ Cf [L,B,H], fp32
// Workspace (~68.2 MB):
//   Wx_bf16 | Wh_bf16 | X_bf16 | hbuf[L][2][B*H] bf16 | C[L][B*H] f32
// ---------------------------------------------------------------------------
extern "C" void kernel_launch(void* const* d_in, const int* in_sizes, int n_in,
                              void* d_out, int out_size, void* d_ws, size_t ws_size,
                              hipStream_t stream) {
    (void)in_sizes; (void)n_in; (void)out_size; (void)ws_size;

    const float* X  = (const float*)d_in[0];
    const float* Wx = (const float*)d_in[1];
    const float* Wh = (const float*)d_in[2];
    const float* bb = (const float*)d_in[3];
    float* out = (float*)d_out;

    const long nWx = (long)NLAYERS * 4 * HID * DIM;   // 8,388,608
    const long nWh = (long)NLAYERS * 4 * HID * HID;   // 8,388,608
    const long nX  = (long)T_STEPS * BATCH * DIM;     // 16,777,216
    const long nBH = (long)BATCH * HID;               // 65,536

    unsigned short* Wx_bf = (unsigned short*)d_ws;
    unsigned short* Wh_bf = Wx_bf + nWx;
    unsigned short* X_bf  = Wh_bf + nWh;
    unsigned short* hbuf  = X_bf + nX;                 // [L][2][B*H] bf16
    float* Cstate = (float*)(hbuf + (long)NLAYERS * 2 * nBH);  // [L][B*H]

    cvt_f32_bf16<<<4096, 256, 0, stream>>>(Wx, Wx_bf, nWx);
    cvt_f32_bf16<<<4096, 256, 0, stream>>>(Wh, Wh_bf, nWh);
    cvt_f32_bf16<<<4096, 256, 0, stream>>>(X,  X_bf,  nX);
    (void)hipMemsetAsync(hbuf,   0, (size_t)NLAYERS * 2 * nBH * sizeof(unsigned short), stream);
    (void)hipMemsetAsync(Cstate, 0, (size_t)NLAYERS * nBH * sizeof(float), stream);

    float* outputs = out;                              // [T][B][H]
    float* Hf = out + (long)T_STEPS * nBH;             // [L][B][H]
    float* Cf = Hf + (long)NLAYERS * nBH;              // [L][B][H]

    for (int t = 0; t < T_STEPS; ++t) {
        const int cur = t & 1, nxt = cur ^ 1;
        const bool last = (t == T_STEPS - 1);

        // layer 0: x = X_bf[t]
        lstm_step<<<64, 256, 0, stream>>>(
            X_bf + (long)t * nBH,
            hbuf + (0 * 2 + cur) * nBH, hbuf + (0 * 2 + nxt) * nBH,
            Wx_bf, Wh_bf, bb, Cstate,
            nullptr,
            last ? Hf : nullptr, last ? Cf : nullptr);

        // layer 1: x = layer0's freshly written h (bf16)
        lstm_step<<<64, 256, 0, stream>>>(
            hbuf + (0 * 2 + nxt) * nBH,
            hbuf + (1 * 2 + cur) * nBH, hbuf + (1 * 2 + nxt) * nBH,
            Wx_bf + (long)4 * HID * DIM, Wh_bf + (long)4 * HID * HID,
            bb + 4 * HID, Cstate + nBH,
            outputs + (long)t * nBH,
            last ? Hf + nBH : nullptr, last ? Cf + nBH : nullptr);
    }
}
